// CausalSelfAttention_30262339568318
// MI455X (gfx1250) — matrix operand order repaired and checked by reading, not compiled, not run on hardware
//
#include <hip/hip_runtime.h>

typedef __attribute__((ext_vector_type(16))) _Float16     v16h;
typedef __attribute__((ext_vector_type(8)))  _Float16     v8h;
typedef __attribute__((ext_vector_type(8)))  float        v8f;
typedef __attribute__((ext_vector_type(4)))  unsigned int v4u;
typedef __attribute__((ext_vector_type(8)))  int          v8i;
typedef __attribute__((ext_vector_type(4)))  int          v4i;

#define BDIM  2
#define TDIM  2048
#define CDIM  1024
#define NHEAD 16
#define HDIM  64

union V16H { v16h v; v8h h[2]; };

// ---------------------------------------------------------------------------
// f32 -> f16 convert
// ---------------------------------------------------------------------------
__global__ void cvt_f32_f16(const float* __restrict__ in,
                            _Float16* __restrict__ out, int n) {
  int i = blockIdx.x * 256 + threadIdx.x;
  if (i < n) out[i] = (_Float16)in[i];
}

// ---------------------------------------------------------------------------
// TDM: issue a 2D f16 tile load (tile_h x tile_w elems) global -> LDS.
// D# built per CDNA5 ISA 8.3/8.4: group0 = {count, lds_addr, global_addr,
// type=2}; group1 = {data_size=2B, tensor_dim0/1, tile_dim0/1, dim0 stride}.
// Groups 2/3 (and the trailing group) zero: 2D tensor.  Tracked by TENSORcnt.
// This toolchain's builtin is the 6-arg form.
// ---------------------------------------------------------------------------
__device__ __forceinline__ void tdm_load_2d_f16(
    unsigned lds_off_bytes, const _Float16* gptr,
    unsigned tensor_w, unsigned tensor_h, unsigned row_stride_elems,
    unsigned tile_w, unsigned tile_h) {
  unsigned long long ga = (unsigned long long)(uintptr_t)gptr;
  v4u g0;
  g0.x = 1u;                                        // count=1, user mode
  g0.y = lds_off_bytes;                             // lds_addr (bytes)
  g0.z = (unsigned)(ga & 0xFFFFFFFFu);              // global_addr[31:0]
  g0.w = (unsigned)((ga >> 32) & 0x01FFFFFFu)       // global_addr[56:32]
         | (2u << 30);                              // type=2 ("image")
  v8i g1;
  g1[0] = (int)(1u << 16);                          // data_size=1 (2 bytes)
  g1[1] = (int)(tensor_w << 16);                    // tensor_dim0[15:0]
  g1[2] = (int)((tensor_w >> 16) | (tensor_h << 16));   // dim0 hi | dim1 lo
  g1[3] = (int)((tensor_h >> 16) | (tile_w << 16));     // dim1 hi | tile_dim0
  g1[4] = (int)tile_h;                              // tile_dim1 (tile_dim2=0)
  g1[5] = (int)row_stride_elems;                    // tensor_dim0_stride[31:0]
  g1[6] = 0;                                        // stride hi | dim1_stride lo
  g1[7] = 0;
  v4i z4 = {0, 0, 0, 0};
  v8i z8 = {0, 0, 0, 0, 0, 0, 0, 0};
  __builtin_amdgcn_tensor_load_to_lds(g0, g1, z4, z4, z8, 0);
}

// ---------------------------------------------------------------------------
// WMMA GEMM: Y[m,n] = sum_k A[m,k] * W[n,k] + bias[n]
// A: [M][K] row-major f16.  W: [N][K] row-major f16 (B-matrix = W^T).
// Each wave computes a 32x32 output block (2x2 tiles); block = 4 waves =
// 64x64.  Whole f16 working set (~48MB) is L2-resident (192MB), so fragments
// come straight from global.
//
// MODE 0: store f16 as [B,H,T,D]           (Q and K)
// MODE 2: store f16 as [B,H,D,T]           (V, pre-transposed for P*V)
// MODE 3: store f32 flat [M][N] with bias  (final projection -> d_out)
// ---------------------------------------------------------------------------
template <int MODE>
__global__ __launch_bounds__(128) void gemm_ws(
    const _Float16* __restrict__ A,
    const _Float16* __restrict__ W,
    const float* __restrict__ bias,
    void* __restrict__ outp, int M, int K) {
  const int lane = threadIdx.x & 31;
  const int w    = threadIdx.x >> 5;
  const int lo = lane & 15, hi = lane >> 4;
  const int mbase = blockIdx.y * 64 + (w >> 1) * 32;
  const int nbase = blockIdx.x * 64 + (w & 1) * 32;

  v8f acc[2][2] = {};

  for (int kb = 0; kb < K; kb += 32) {
    V16H a[2], b[2];
#pragma unroll
    for (int mt = 0; mt < 2; ++mt) {
      const _Float16* ap = A + (size_t)(mbase + mt * 16 + lo) * K + kb + 8 * hi;
      a[mt].h[0] = *(const v8h*)(ap);
      a[mt].h[1] = *(const v8h*)(ap + 16);
      if (kb + 32 < K) __builtin_prefetch(ap + 32, 0, 0);
    }
#pragma unroll
    for (int nt = 0; nt < 2; ++nt) {
      const _Float16* bp = W + (size_t)(nbase + nt * 16 + lo) * K + kb + 8 * hi;
      b[nt].h[0] = *(const v8h*)(bp);
      b[nt].h[1] = *(const v8h*)(bp + 16);
      if (kb + 32 < K) __builtin_prefetch(bp + 32, 0, 0);
    }
#pragma unroll
    for (int mt = 0; mt < 2; ++mt)
#pragma unroll
      for (int nt = 0; nt < 2; ++nt)
        acc[mt][nt] = __builtin_amdgcn_wmma_f32_16x16x32_f16(
            false, a[mt].v, false, b[nt].v, (short)0, acc[mt][nt], false, false);
  }

#pragma unroll
  for (int mt = 0; mt < 2; ++mt)
#pragma unroll
    for (int nt = 0; nt < 2; ++nt)
#pragma unroll
      for (int r = 0; r < 8; ++r) {
        int m = mbase + mt * 16 + r + 8 * hi;
        int n = nbase + nt * 16 + lo;
        float val = acc[mt][nt][r] + bias[n];
        if (MODE == 3) {
          ((float*)outp)[(size_t)m * CDIM + n] = val;
        } else {
          int bb = m >> 11;           // m = b*T + t
          int t  = m & (TDIM - 1);
          int h  = n >> 6;            // n = h*64 + d
          int d  = n & 63;
          _Float16* o = (_Float16*)outp;
          if (MODE == 2)
            o[(((size_t)bb * NHEAD + h) * HDIM + d) * TDIM + t] = (_Float16)val;
          else
            o[(((size_t)bb * NHEAD + h) * TDIM + t) * HDIM + d] = (_Float16)val;
        }
      }
}

// ---------------------------------------------------------------------------
// Flash attention (causal, online softmax), TDM-staged K/V.
// Grid: (T/64, B*H).  Block: 128 threads = 4 waves; each wave owns 16 q-rows.
// Wave 0 double-buffers 64x64 K and V tiles into LDS with tensor_load_to_lds
// while all 4 waves run WMMAs on the previous tile; K/V are fetched from L2
// once per block instead of once per wave.
// Q,K: [B,H,T,D] f16.  V: [B,H,D,T] f16 (transposed).  Y: [B,T,C] f16.
// ---------------------------------------------------------------------------
__global__ __launch_bounds__(128) void flash_attn(
    const _Float16* __restrict__ Q,
    const _Float16* __restrict__ Km,
    const _Float16* __restrict__ Vt,
    _Float16* __restrict__ Y) {
  __shared__ __align__(16) _Float16 kbuf[2][64][64];   // [buf][kv][d]
  __shared__ __align__(16) _Float16 vbuf[2][64][64];   // [buf][d][kv]
  __shared__ __align__(16) _Float16 pbuf[4][16][64];   // per-wave P staging

  const int lane = threadIdx.x & 31;
  const int w = threadIdx.x >> 5;
  const int lo = lane & 15, hi = lane >> 4;
  const int qb = blockIdx.x;
  const int bh = blockIdx.y;
  const int b = bh / NHEAD, h = bh % NHEAD;

  const _Float16* Qbh = Q  + (size_t)bh * TDIM * HDIM;
  const _Float16* Kbh = Km + (size_t)bh * TDIM * HDIM;
  const _Float16* Vbh = Vt + (size_t)bh * HDIM * TDIM;

  const int mrow = qb * 64 + w * 16;

  // Q fragments (K-dim = 64 -> two x32 steps), loaded once
  V16H qf[2];
#pragma unroll
  for (int ks = 0; ks < 2; ++ks) {
    const _Float16* qp = Qbh + (size_t)(mrow + lo) * HDIM + ks * 32 + 8 * hi;
    qf[ks].h[0] = *(const v8h*)(qp);
    qf[ks].h[1] = *(const v8h*)(qp + 16);
  }

  // TDM issue for kv-tile `kt` into buffer `nb`
  auto issue_tile = [&](int kt, int nb) {
    unsigned koff = (unsigned)(uintptr_t)(void*)&kbuf[nb][0][0];  // LDS offset
    unsigned voff = (unsigned)(uintptr_t)(void*)&vbuf[nb][0][0];
    // K tile: rows kv=kt*64..+63 of [T][D] -> LDS [64][64]
    tdm_load_2d_f16(koff, Kbh + (size_t)kt * 64 * HDIM,
                    HDIM, (unsigned)(TDIM - kt * 64), HDIM, HDIM, 64);
    // V tile: 64 d-rows, cols kv=kt*64..+63 of [D][T] (stride T) -> LDS [64][64]
    tdm_load_2d_f16(voff, Vbh + (size_t)kt * 64,
                    (unsigned)(TDIM - kt * 64), HDIM, TDIM, 64, HDIM);
  };

  if (w == 0) issue_tile(0, 0);

  v8f oacc[4] = {};
  float rm[8], rl[8];
#pragma unroll
  for (int r = 0; r < 8; ++r) { rm[r] = -1e30f; rl[r] = 0.f; }

  for (int kt = 0; kt <= qb; ++kt) {
    const int cur = kt & 1;
    if (w == 0) {
      if (kt + 1 <= qb) {
        issue_tile(kt + 1, cur ^ 1);                    // overlap next DMA
        __builtin_amdgcn_s_wait_tensorcnt((short)2);    // current tile done
      } else {
        __builtin_amdgcn_s_wait_tensorcnt((short)0);
      }
    }
    __syncthreads();  // K/V tile visible to all waves

    // ---- S = (Q K^T) * 1/sqrt(D), K fragments from LDS --------------------
    v8f s[4] = {};
#pragma unroll
    for (int nt = 0; nt < 4; ++nt) {
#pragma unroll
      for (int ks = 0; ks < 2; ++ks) {
        V16H kf;
        const _Float16* kp = &kbuf[cur][nt * 16 + lo][ks * 32 + 8 * hi];
        kf.h[0] = *(const v8h*)(kp);
        kf.h[1] = *(const v8h*)(kp + 16);
        s[nt] = __builtin_amdgcn_wmma_f32_16x16x32_f16(
            false, qf[ks].v, false, kf.v, (short)0, s[nt], false, false);
      }
    }

    const bool diag = (kt == qb);
#pragma unroll
    for (int nt = 0; nt < 4; ++nt)
#pragma unroll
      for (int r = 0; r < 8; ++r) {
        float x = s[nt][r] * 0.125f;  // 1/sqrt(64)
        if (diag) {
          int tq = mrow + r + 8 * hi;
          int tk = kt * 64 + nt * 16 + lo;
          if (tk > tq) x = -1e30f;
        }
        s[nt][r] = x;
      }

    // ---- online softmax (rows live across 16-lane groups) -----------------
#pragma unroll
    for (int r = 0; r < 8; ++r) {
      float tmax = fmaxf(fmaxf(s[0][r], s[1][r]), fmaxf(s[2][r], s[3][r]));
#pragma unroll
      for (int m = 1; m <= 8; m <<= 1) tmax = fmaxf(tmax, __shfl_xor(tmax, m, 32));
      float nm = fmaxf(rm[r], tmax);
      float corr = __expf(rm[r] - nm);
      rm[r] = nm;
      float tsum = 0.f;
#pragma unroll
      for (int nt = 0; nt < 4; ++nt) {
        float p = __expf(s[nt][r] - nm);
        s[nt][r] = p;
        tsum += p;
      }
#pragma unroll
      for (int m = 1; m <= 8; m <<= 1) tsum += __shfl_xor(tsum, m, 32);
      rl[r] = rl[r] * corr + tsum;
#pragma unroll
      for (int dt = 0; dt < 4; ++dt) oacc[dt][r] *= corr;
    }

    // ---- P: accumulator (C) layout -> A-fragment layout via LDS -----------
#pragma unroll
    for (int nt = 0; nt < 4; ++nt)
#pragma unroll
      for (int r = 0; r < 8; ++r)
        pbuf[w][r + 8 * hi][nt * 16 + lo] = (_Float16)s[nt][r];
    __syncthreads();

    // ---- O += P * V, V fragments from LDS ---------------------------------
#pragma unroll
    for (int ks = 0; ks < 2; ++ks) {
      V16H pa;
      const _Float16* pp = &pbuf[w][lo][ks * 32 + 8 * hi];
      pa.h[0] = *(const v8h*)(pp);
      pa.h[1] = *(const v8h*)(pp + 16);
#pragma unroll
      for (int dt = 0; dt < 4; ++dt) {
        V16H vf;
        const _Float16* vp = &vbuf[cur][dt * 16 + lo][ks * 32 + 8 * hi];
        vf.h[0] = *(const v8h*)(vp);
        vf.h[1] = *(const v8h*)(vp + 16);
        oacc[dt] = __builtin_amdgcn_wmma_f32_16x16x32_f16(
            false, pa.v, false, vf.v, (short)0, oacc[dt], false, false);
      }
    }
    __syncthreads();  // all waves done reading tile before TDM overwrites it
  }

  // ---- normalize and store to flat [B,T,C] --------------------------------
#pragma unroll
  for (int dt = 0; dt < 4; ++dt)
#pragma unroll
    for (int r = 0; r < 8; ++r) {
      int tq = mrow + r + 8 * hi;
      int d = dt * 16 + lo;
      float val = oacc[dt][r] / rl[r];
      Y[((size_t)b * TDIM + tq) * CDIM + h * HDIM + d] = (_Float16)val;
    }
}

// ---------------------------------------------------------------------------
// Launch
// ---------------------------------------------------------------------------
extern "C" void kernel_launch(void* const* d_in, const int* in_sizes, int n_in,
                              void* d_out, int out_size, void* d_ws,
                              size_t ws_size, hipStream_t stream) {
  const float* x  = (const float*)d_in[0];
  const float* Wq = (const float*)d_in[1];
  const float* bq = (const float*)d_in[2];
  const float* Wk = (const float*)d_in[3];
  const float* bk = (const float*)d_in[4];
  const float* Wv = (const float*)d_in[5];
  const float* bv = (const float*)d_in[6];
  const float* Wo = (const float*)d_in[7];
  const float* bo = (const float*)d_in[8];
  float* out = (float*)d_out;

  const size_t Mrows = (size_t)BDIM * TDIM;  // 4096

  _Float16* ws  = (_Float16*)d_ws;
  _Float16* xh  = ws;                         // [M][C]
  _Float16* wqh = xh  + Mrows * CDIM;
  _Float16* wkh = wqh + (size_t)CDIM * CDIM;
  _Float16* wvh = wkh + (size_t)CDIM * CDIM;
  _Float16* woh = wvh + (size_t)CDIM * CDIM;
  _Float16* qh  = woh + (size_t)CDIM * CDIM;  // [B,H,T,D]
  _Float16* kh  = qh  + Mrows * CDIM;         // [B,H,T,D]
  _Float16* vth = kh  + Mrows * CDIM;         // [B,H,D,T]
  _Float16* yh  = vth + Mrows * CDIM;         // [B,T,C]

  int nx = (int)(Mrows * CDIM);
  int nw = CDIM * CDIM;
  cvt_f32_f16<<<(nx + 255) / 256, 256, 0, stream>>>(x,  xh,  nx);
  cvt_f32_f16<<<(nw + 255) / 256, 256, 0, stream>>>(Wq, wqh, nw);
  cvt_f32_f16<<<(nw + 255) / 256, 256, 0, stream>>>(Wk, wkh, nw);
  cvt_f32_f16<<<(nw + 255) / 256, 256, 0, stream>>>(Wv, wvh, nw);
  cvt_f32_f16<<<(nw + 255) / 256, 256, 0, stream>>>(Wo, woh, nw);

  dim3 gb(CDIM / 64, (unsigned)(Mrows / 64));
  gemm_ws<0><<<gb, 128, 0, stream>>>(xh, wqh, bq, qh,  (int)Mrows, CDIM);
  gemm_ws<0><<<gb, 128, 0, stream>>>(xh, wkh, bk, kh,  (int)Mrows, CDIM);
  gemm_ws<2><<<gb, 128, 0, stream>>>(xh, wvh, bv, vth, (int)Mrows, CDIM);

  flash_attn<<<dim3(TDIM / 64, BDIM * NHEAD), 128, 0, stream>>>(qh, kh, vth, yh);

  gemm_ws<3><<<gb, 128, 0, stream>>>(yh, woh, bo, out, (int)Mrows, CDIM);
}
